// ANIModelShare_34385508172240
// MI455X (gfx1250) — matrix operand order repaired and checked by reading, not compiled, not run on hardware
//
#include <hip/hip_runtime.h>
#include <hip/hip_bf16.h>

typedef __attribute__((ext_vector_type(16))) __bf16 v16bf;
typedef __attribute__((ext_vector_type(8)))  __bf16 v8bf;
typedef __attribute__((ext_vector_type(8)))  float  v8f;

#define D_IN  1008
#define H0    768
#define H1_   192
#define H2_   96
#define N_ATOMS (512*96)

#define WMMA_BF16(a, b, c) \
    __builtin_amdgcn_wmma_f32_16x16x32_bf16(false, (a), false, (b), (short)0, (c), false, false)

__device__ __forceinline__ float celu01(float x) {
    return x > 0.0f ? x : 0.1f * (__expf(x * 10.0f) - 1.0f);
}

// A-matrix fragment (16x32 bf16), lds row-major [m][k], k-contiguous.
// lane 0-15: M=lane, halves 0..7 -> K=g*8+0..7, halves 8..15 -> K=16+g*8+0..7 (g = lane>>4)
__device__ __forceinline__ v16bf frag_A(const __bf16* lds, int stride, int row0, int k0, int lane) {
    int r = row0 + (lane & 15);
    int g = lane >> 4;
    const __bf16* p = lds + r * stride + k0 + g * 8;
    v8bf lo = *(const v8bf*)p;
    v8bf hi = *(const v8bf*)(p + 16);
    return __builtin_shufflevector(lo, hi, 0,1,2,3,4,5,6,7,8,9,10,11,12,13,14,15);
}

// B-matrix fragment (32x16 bf16), lds stored TRANSPOSED [n][k], k-contiguous.
// lane 0-15: N=lane, halves 0..7 -> K=g*8+0..7, halves 8..15 -> K=16+g*8+0..7 (g = lane>>4)
__device__ __forceinline__ v16bf frag_B(const __bf16* lds, int stride, int col0, int k0, int lane) {
    int c = col0 + (lane & 15);
    int g = lane >> 4;
    const __bf16* p = lds + c * stride + k0 + g * 8;
    v8bf lo = *(const v8bf*)p;
    v8bf hi = *(const v8bf*)(p + 16);
    return __builtin_shufflevector(lo, hi, 0,1,2,3,4,5,6,7,8,9,10,11,12,13,14,15);
}

// -------- Kernel 1: h = celu(aev @ Ws + bs)  [49152,1008]x[1008,768] -> bf16 --------
// 128x64 block tile, 8 waves each 32x32 (2x2 WMMA), K padded 1008->1024,
// software-pipelined: global->regs for step s+1 overlapped with WMMA on LDS buf p.
__global__ __launch_bounds__(256) void k_gemm1(const float* __restrict__ aev,
                                               const float* __restrict__ Ws,
                                               const float* __restrict__ bs,
                                               __bf16* __restrict__ h) {
    __shared__ __bf16 Ah[2][128 * 40];   // [128 m][32 k + 8 pad]
    __shared__ __bf16 Bw[2][64 * 40];    // [64 n][32 k + 8 pad], transposed

    const int tid   = threadIdx.x;
    const int lane  = tid & 31;
    const int wid   = tid >> 5;
    const int waveM = wid & 3;        // 4 waves along M (32 rows each)
    const int waveN = wid >> 2;       // 2 waves along N (32 cols each)
    const int nBase = (blockIdx.x % 12) * 64;
    const int mBase = (blockIdx.x / 12) * 128;

    // per-thread fill coordinates (constant across K-steps)
    const int ar_r = tid >> 3;            // A row 0..31 (+ i*32)
    const int ar_c = (tid & 7) * 4;       // A k-offset within 32-wide tile
    const int br_r = tid >> 4;            // B k-row 0..15 (+ i*16)
    const int br_c = (tid & 15) * 4;      // B n-offset

    float4 ar[4], br[2];

    auto stage = [&](int s) {             // global -> registers for K-step s
        const int k0 = s * 32;
        if (s < 31) {                     // fully in-bounds: unconditional b128 loads
            #pragma unroll
            for (int i = 0; i < 4; i++)
                ar[i] = *(const float4*)(aev + (size_t)(mBase + ar_r + i * 32) * D_IN + k0 + ar_c);
            #pragma unroll
            for (int i = 0; i < 2; i++)
                br[i] = *(const float4*)(Ws + (size_t)(k0 + br_r + i * 16) * H0 + nBase + br_c);
        } else {                          // tail step 992..1023: guard + zero pad
            #pragma unroll
            for (int i = 0; i < 4; i++) {
                float4 v = make_float4(0.f, 0.f, 0.f, 0.f);
                if (k0 + ar_c < D_IN)
                    v = *(const float4*)(aev + (size_t)(mBase + ar_r + i * 32) * D_IN + k0 + ar_c);
                ar[i] = v;
            }
            #pragma unroll
            for (int i = 0; i < 2; i++) {
                float4 v = make_float4(0.f, 0.f, 0.f, 0.f);
                if (k0 + br_r + i * 16 < D_IN)
                    v = *(const float4*)(Ws + (size_t)(k0 + br_r + i * 16) * H0 + nBase + br_c);
                br[i] = v;
            }
        }
    };
    auto commit = [&](int p) {            // registers -> LDS buffer p (f32 -> bf16)
        #pragma unroll
        for (int i = 0; i < 4; i++) {
            __bf16* d = &Ah[p][(ar_r + i * 32) * 40 + ar_c];
            d[0] = (__bf16)ar[i].x; d[1] = (__bf16)ar[i].y;
            d[2] = (__bf16)ar[i].z; d[3] = (__bf16)ar[i].w;
        }
        #pragma unroll
        for (int i = 0; i < 2; i++) {
            int r = br_r + i * 16;
            Bw[p][(br_c + 0) * 40 + r] = (__bf16)br[i].x;
            Bw[p][(br_c + 1) * 40 + r] = (__bf16)br[i].y;
            Bw[p][(br_c + 2) * 40 + r] = (__bf16)br[i].z;
            Bw[p][(br_c + 3) * 40 + r] = (__bf16)br[i].w;
        }
    };

    v8f acc[2][2];
    v8f zero = {};
    acc[0][0] = zero; acc[0][1] = zero; acc[1][0] = zero; acc[1][1] = zero;

    stage(0);
    commit(0);
    int p = 0;
    for (int s = 0; s < 32; s++) {
        __syncthreads();
        if (s + 1 < 32) stage(s + 1);   // issue next tile's global loads early
        v16bf a0 = frag_A(Ah[p], 40, waveM * 32 + 0,  0, lane);
        v16bf a1 = frag_A(Ah[p], 40, waveM * 32 + 16, 0, lane);
        v16bf b0 = frag_B(Bw[p], 40, waveN * 32 + 0,  0, lane);
        v16bf b1 = frag_B(Bw[p], 40, waveN * 32 + 16, 0, lane);
        acc[0][0] = WMMA_BF16(a0, b0, acc[0][0]);
        acc[0][1] = WMMA_BF16(a0, b1, acc[0][1]);
        acc[1][0] = WMMA_BF16(a1, b0, acc[1][0]);
        acc[1][1] = WMMA_BF16(a1, b1, acc[1][1]);
        if (s + 1 < 32) commit(p ^ 1);  // write next tile into the other buffer
        p ^= 1;
    }
    // epilogue: bias + CELU, store bf16. C layout: VGPR v -> M = v + 8*(lane>>4), N = lane&15
    const int lg = lane >> 4, ln = lane & 15;
    for (int j = 0; j < 2; j++) {
        int col = nBase + waveN * 32 + j * 16 + ln;
        float bias = bs[col];
        for (int i = 0; i < 2; i++) {
            int rb = mBase + waveM * 32 + i * 16 + 8 * lg;
            for (int v = 0; v < 8; v++) {
                float val = celu01(acc[i][j][v] + bias);
                h[(size_t)(rb + v) * H0 + col] = (__bf16)val;
            }
        }
    }
}

// -------- Kernel 2: fused per-species MLPs + routing, 32 atoms per workgroup --------
// smem layout (bytes):
//   hT    [32][776] bf16 : 0      .. 49664
//   h1T   [32][200] bf16 : 49664  .. 62464
//   h2T   [32][104] bf16 : 62464  .. 69120
//   bufB  [192][72] bf16 : 69120  .. 96768   (W1 k-chunk, transposed)
//   bufB2 [96][200] bf16 : 96768  .. 135168  (W2 full, transposed)
//   pbuf  [32][8]   f32  : 135168 .. 136192
//   eT    [32]      f32  : 136192 .. 136320
//   spT   [32]      i32  : 136320 .. 136448
#define K2_SMEM 136448

__global__ __launch_bounds__(256) void k_mlp2(const __bf16* __restrict__ h,
                                              const int*    __restrict__ species,
                                              const float* __restrict__ W1, const float* __restrict__ b1,
                                              const float* __restrict__ W2, const float* __restrict__ b2,
                                              const float* __restrict__ W3, const float* __restrict__ b3,
                                              float* __restrict__ eOut) {
    extern __shared__ char smem[];
    __bf16* hT    = (__bf16*)(smem);
    __bf16* h1T   = (__bf16*)(smem + 49664);
    __bf16* h2T   = (__bf16*)(smem + 62464);
    __bf16* bufB  = (__bf16*)(smem + 69120);
    __bf16* bufB2 = (__bf16*)(smem + 96768);
    float*  pbuf  = (float*)(smem + 135168);
    float*  eT    = (float*)(smem + 136192);
    int*    spT   = (int*)(smem + 136320);

    const int tid  = threadIdx.x;
    const int lane = tid & 31;
    const int wid  = tid >> 5;
    const int atomBase = blockIdx.x * 32;
    const int lg = lane >> 4, ln = lane & 15;
    const int mi = wid & 1;   // M sub-tile (constant per wave in both GEMMs)

    // stage h tile [32][768] bf16 into LDS (16B chunks)
    for (int q = tid; q < 3072; q += 256) {
        int r = q / 96, c8 = q % 96;
        *(uint4*)(hT + r * 776 + c8 * 8) =
            *(const uint4*)(h + (size_t)(atomBase + r) * H0 + c8 * 8);
    }
    if (tid < 32) {
        spT[tid] = species[atomBase + tid];
        eT[tid] = 0.0f;
    }
    __syncthreads();

    v8f zero = {};
    for (int s = 0; s < 4; s++) {
        // ---- GEMM A: h[32,768] @ W1[s][768,192] -> h1, CELU ----
        v8f acc[3];
        acc[0] = zero; acc[1] = zero; acc[2] = zero;
        for (int kc = 0; kc < 12; kc++) {          // K chunks of 64
            __syncthreads();
            for (int q = tid; q < 3072; q += 256) { // W1 chunk 64x192 -> [n][k]
                int r = q / 48, c4 = q % 48;
                float4 v = *(const float4*)(W1 + ((size_t)s * H0 + kc * 64 + r) * H1_ + c4 * 4);
                bufB[(c4 * 4 + 0) * 72 + r] = (__bf16)v.x;
                bufB[(c4 * 4 + 1) * 72 + r] = (__bf16)v.y;
                bufB[(c4 * 4 + 2) * 72 + r] = (__bf16)v.z;
                bufB[(c4 * 4 + 3) * 72 + r] = (__bf16)v.w;
            }
            __syncthreads();
            for (int kk = 0; kk < 2; kk++) {
                int kg = kc * 64 + kk * 32;
                v16bf a = frag_A(hT, 776, mi * 16, kg, lane);
                for (int t3 = 0; t3 < 3; t3++) {
                    int ni = (wid >> 1) + t3 * 4;
                    v16bf b = frag_B(bufB, 72, ni * 16, kk * 32, lane);
                    acc[t3] = WMMA_BF16(a, b, acc[t3]);
                }
            }
        }
        __syncthreads();
        for (int t3 = 0; t3 < 3; t3++) {
            int ni = (wid >> 1) + t3 * 4;
            int col = ni * 16 + ln;
            float bias = b1[s * H1_ + col];
            int rb = mi * 16 + 8 * lg;
            for (int v = 0; v < 8; v++)
                h1T[(rb + v) * 200 + col] = (__bf16)celu01(acc[t3][v] + bias);
        }
        // W2[s] 192x96 -> [n][k] transposed
        for (int q = tid; q < 4608; q += 256) {
            int r = q / 24, c4 = q % 24;
            float4 v = *(const float4*)(W2 + ((size_t)s * H1_ + r) * H2_ + c4 * 4);
            bufB2[(c4 * 4 + 0) * 200 + r] = (__bf16)v.x;
            bufB2[(c4 * 4 + 1) * 200 + r] = (__bf16)v.y;
            bufB2[(c4 * 4 + 2) * 200 + r] = (__bf16)v.z;
            bufB2[(c4 * 4 + 3) * 200 + r] = (__bf16)v.w;
        }
        __syncthreads();
        // ---- GEMM B: h1[32,192] @ W2[s][192,96] -> h2, CELU ----
        {
            int nacc = (wid < 4) ? 2 : 1;           // 12 tiles over 8 waves
            v8f acc2[2];
            acc2[0] = zero; acc2[1] = zero;
            for (int ks = 0; ks < 6; ks++) {
                v16bf a = frag_A(h1T, 200, mi * 16, ks * 32, lane);
                for (int t2 = 0; t2 < nacc; t2++) {
                    int ni = (wid + t2 * 8) >> 1;
                    v16bf b = frag_B(bufB2, 200, ni * 16, ks * 32, lane);
                    acc2[t2] = WMMA_BF16(a, b, acc2[t2]);
                }
            }
            for (int t2 = 0; t2 < nacc; t2++) {
                int ni = (wid + t2 * 8) >> 1;
                int col = ni * 16 + ln;
                float bias = b2[s * H2_ + col];
                int rb = mi * 16 + 8 * lg;
                for (int v = 0; v < 8; v++)
                    h2T[(rb + v) * 104 + col] = (__bf16)celu01(acc2[t2][v] + bias);
            }
        }
        __syncthreads();
        // ---- e = h2 . W3[s] + b3[s]; route by species ----
        {
            int m = tid >> 3, sub = tid & 7;
            float p = 0.0f;
            for (int j = 0; j < 12; j++) {
                int k = sub * 12 + j;
                p += (float)h2T[m * 104 + k] * W3[s * H2_ + k];
            }
            pbuf[m * 8 + sub] = p;
        }
        __syncthreads();
        if (tid < 32) {
            float tot = b3[s];
            for (int j = 0; j < 8; j++) tot += pbuf[tid * 8 + j];
            int sp = spT[tid];
            int idx = sp < 0 ? 0 : (sp > 3 ? 3 : sp);
            if (sp >= 0 && idx == s) eT[tid] = tot;
        }
        __syncthreads();
    }
    if (tid < 32) eOut[atomBase + tid] = eT[tid];
}

// -------- Kernel 3: deterministic per-molecule reduction (96 atoms, 1 wave/molecule) --------
__global__ __launch_bounds__(256) void k_reduce(const float* __restrict__ e,
                                                float* __restrict__ out) {
    int wid = threadIdx.x >> 5, lane = threadIdx.x & 31;
    int mol = blockIdx.x * 8 + wid;
    const float* p = e + (size_t)mol * 96;
    float v = p[lane] + p[lane + 32] + p[lane + 64];
    v += __shfl_down(v, 16, 32);
    v += __shfl_down(v, 8, 32);
    v += __shfl_down(v, 4, 32);
    v += __shfl_down(v, 2, 32);
    v += __shfl_down(v, 1, 32);
    if (lane == 0) out[mol] = v;
}

extern "C" void kernel_launch(void* const* d_in, const int* in_sizes, int n_in,
                              void* d_out, int out_size, void* d_ws, size_t ws_size,
                              hipStream_t stream) {
    const int*   species = (const int*)d_in[0];
    const float* aev = (const float*)d_in[1];
    const float* Ws  = (const float*)d_in[2];
    const float* bs  = (const float*)d_in[3];
    const float* W1  = (const float*)d_in[4];
    const float* b1  = (const float*)d_in[5];
    const float* W2  = (const float*)d_in[6];
    const float* b2  = (const float*)d_in[7];
    const float* W3  = (const float*)d_in[8];
    const float* b3  = (const float*)d_in[9];

    __bf16* hbuf = (__bf16*)d_ws;                                    // 49152*768 bf16
    float*  ebuf = (float*)((char*)d_ws + (size_t)N_ATOMS * H0 * 2); // 49152 f32

    k_gemm1<<<dim3((N_ATOMS / 128) * 12), dim3(256), 0, stream>>>(aev, Ws, bs, hbuf);
    k_mlp2<<<dim3(N_ATOMS / 32), dim3(256), K2_SMEM, stream>>>(hbuf, species,
                                                               W1, b1, W2, b2, W3, b3, ebuf);
    k_reduce<<<dim3(64), dim3(256), 0, stream>>>(ebuf, (float*)d_out);
    (void)in_sizes; (void)n_in; (void)out_size; (void)ws_size;
}
